// OverlappingCrossAttention_61134564491470
// MI455X (gfx1250) — compile-verified, matrix operand order checked by reading, not run on hardware
//
#include <hip/hip_runtime.h>
#include <hip/hip_bf16.h>

typedef __attribute__((ext_vector_type(16))) __bf16 bf16x16;
typedef __attribute__((ext_vector_type(8)))  float  f32x8;
typedef __attribute__((ext_vector_type(4)))  int    v4i_t;

#define DIM     384
#define NHEADS  12
#define HD      32
#define SHIFT   4
#define BATCH   4
#define MROWS   (BATCH * 256 * 256)   // 262144

// ---------------------------------------------------------------------------
// Optional CDNA5 async global->LDS path (guarded; falls back to sync copies)
// ---------------------------------------------------------------------------
#if defined(__has_builtin)
#if __has_builtin(__builtin_amdgcn_global_load_async_to_lds_b128) && \
    __has_builtin(__builtin_amdgcn_s_wait_asynccnt)
#define HAVE_ASYNC_LDS 1
#endif
#endif
#ifndef HAVE_ASYNC_LDS
#define HAVE_ASYNC_LDS 0
#endif

// Copy 16 bytes global -> LDS (per-lane addresses). Async when available.
__device__ __forceinline__ void copy16_g2l(const __bf16* g, __bf16* l) {
#if HAVE_ASYNC_LDS
    __builtin_amdgcn_global_load_async_to_lds_b128(
        (__attribute__((address_space(1))) v4i_t*)(void*)g,
        (__attribute__((address_space(3))) v4i_t*)(void*)l, 0, 0);
#else
    *(uint4*)l = *(const uint4*)g;
#endif
}

__device__ __forceinline__ void wait_g2l() {
#if HAVE_ASYNC_LDS
    __builtin_amdgcn_s_wait_asynccnt(0);
#endif
}

// ---------------------------------------------------------------------------
// WMMA helpers
// ---------------------------------------------------------------------------
__device__ __forceinline__ f32x8 wmma_bf16(bf16x16 a, bf16x16 b, f32x8 c) {
    // (neg_a, A, neg_b, B, c_mod, C, reuse_a, reuse_b)
    return __builtin_amdgcn_wmma_f32_16x16x32_bf16(false, a, false, b, (short)0, c,
                                                   false, false);
}

// Load a 16x32 bf16 fragment (A-layout; same gather works for B when the matrix
// is stored N-major/K-contiguous). ISA 16-bit A layout: lanes 0-15 hold
// K = {0..7, 16..23}, lanes 16-31 hold K = {8..15, 24..31}. stride_h is a
// multiple of 8 halves so the two 8-half chunks are 16B-aligned ds_load_b128s.
__device__ __forceinline__ bf16x16 load_frag(const __bf16* base, int row,
                                             int stride_h, int kbase) {
    const int lane = threadIdx.x & 31;
    const __bf16* p = base + (size_t)(row + (lane & 15)) * stride_h + kbase +
                      ((lane >> 4) << 3);
    bf16x16 f;
#pragma unroll
    for (int i = 0; i < 8; ++i) {
        f[i]     = p[i];        // K = k0 .. k0+7
        f[i + 8] = p[i + 16];   // K = k0+16 .. k0+23
    }
    return f;
}

__device__ __forceinline__ unsigned pack2bf(float a, float b) {
    union { __bf16 h[2]; unsigned u; } u;
    u.h[0] = (__bf16)a; u.h[1] = (__bf16)b;
    return u.u;
}

// A tile row: 16 fp32 -> 8 packed bf16 dwords -> two b128 LDS stores.
__device__ __forceinline__ void store_arow_f32(const float* src, unsigned* dst8) {
    const float4* s4 = (const float4*)src;
    float4 f0 = s4[0], f1 = s4[1], f2 = s4[2], f3 = s4[3];
    uint4 o0, o1;
    o0.x = pack2bf(f0.x, f0.y); o0.y = pack2bf(f0.z, f0.w);
    o0.z = pack2bf(f1.x, f1.y); o0.w = pack2bf(f1.z, f1.w);
    o1.x = pack2bf(f2.x, f2.y); o1.y = pack2bf(f2.z, f2.w);
    o1.z = pack2bf(f3.x, f3.y); o1.w = pack2bf(f3.z, f3.w);
    *(uint4*)dst8       = o0;
    *(uint4*)(dst8 + 4) = o1;
}

// B tile (W[k0..k0+31][n0..n0+63] bf16) -> LDS transposed sB[n][k], stride 40
// halves (20 dwords). Thread pairs two K-rows so each store is a packed b32.
__device__ __forceinline__ void load_btile(const __bf16* Wb, int N, int k0, int n0,
                                           unsigned* sB32, int t) {
    int kp   = t >> 3;          // 0..15  (k = 2*kp)
    int nseg = (t & 7) << 3;    // 0,8,...,56
    const __bf16* r0 = Wb + (size_t)(k0 + 2 * kp) * N + n0 + nseg;
    const __bf16* r1 = r0 + N;
    union { uint4 v; unsigned short s[8]; } a, b;
    a.v = *(const uint4*)r0;
    b.v = *(const uint4*)r1;
#pragma unroll
    for (int i = 0; i < 8; ++i)
        sB32[(nseg + i) * 20 + kp] = (unsigned)a.s[i] | ((unsigned)b.s[i] << 16);
}

// ---------------------------------------------------------------------------
// Weight fp32 -> bf16 conversion
// ---------------------------------------------------------------------------
__global__ void cvt_f32_bf16(const float* __restrict__ src, __bf16* __restrict__ dst,
                             int n) {
    int i = blockIdx.x * blockDim.x + threadIdx.x;
    if (i < n) dst[i] = (__bf16)src[i];
}

// ---------------------------------------------------------------------------
// GEMM: Out[M,N](bf16) = bf16(A fp32, optionally rolled rows) @ W(bf16) + bias
// Tile 64x64, 128 threads (4 waves), each wave computes a 16x64 strip.
// ---------------------------------------------------------------------------
template <int SHIFTED>
__global__ __launch_bounds__(128) void gemm_x_bf16(
    const float* __restrict__ A, const __bf16* __restrict__ Wb,
    const float* __restrict__ bias, __bf16* __restrict__ Out, int N) {
    __shared__ unsigned sA32[64 * 20];   // 64 rows x 32 k, stride 40 halves
    __shared__ unsigned sB32[64 * 20];   // N-major (transposed), stride 40 halves
    __bf16* sA = (__bf16*)sA32;
    __bf16* sB = (__bf16*)sB32;

    const int n0 = blockIdx.x * 64;
    const int m0 = blockIdx.y * 64;
    const int t  = threadIdx.x;
    const int w  = t >> 5;
    const int lane = t & 31;

    // A-tile row mapping (roll folded into the load index)
    int arow = m0 + (t >> 1);
    if (SHIFTED) {
        int wc = arow & 255, hc = (arow >> 8) & 255, bc = arow >> 16;
        arow = (bc << 16) | (((hc + SHIFT) & 255) << 8) | ((wc + SHIFT) & 255);
    }
    const int cseg = (t & 1) << 4;

    f32x8 acc[4] = {};

    for (int k0 = 0; k0 < DIM; k0 += 32) {
        __syncthreads();
        store_arow_f32(A + (size_t)arow * DIM + k0 + cseg,
                       &sA32[(t >> 1) * 20 + (cseg >> 1)]);
        load_btile(Wb, N, k0, n0, sB32, t);
        if (k0 + 32 < DIM)  // prefetch next weight tile (global_prefetch_b8)
            __builtin_prefetch(Wb + (size_t)(k0 + 32 + (t >> 3) * 2) * N + n0, 0, 1);
        __syncthreads();

        bf16x16 af = load_frag(sA, w * 16, 40, 0);
#pragma unroll
        for (int tt = 0; tt < 4; ++tt) {
            bf16x16 bfrag = load_frag(sB, tt * 16, 40, 0);
            acc[tt] = wmma_bf16(af, bfrag, acc[tt]);
        }
    }

    // Epilogue: C/D layout: lane holds (m = j + 8*(lane>=16), n = lane%16)
#pragma unroll
    for (int tt = 0; tt < 4; ++tt) {
#pragma unroll
        for (int j = 0; j < 8; ++j) {
            int m   = (w << 4) + j + ((lane >> 4) << 3);
            int col = n0 + (tt << 4) + (lane & 15);
            float v = acc[tt][j] + bias[col];
            Out[(size_t)(m0 + m) * N + col] = (__bf16)v;
        }
    }
}

// ---------------------------------------------------------------------------
// GEMM proj: out[M,384](fp32) = A(bf16) @ W(bf16) + bias
// ---------------------------------------------------------------------------
__global__ __launch_bounds__(128) void gemm_proj(
    const __bf16* __restrict__ A, const __bf16* __restrict__ Wb,
    const float* __restrict__ bias, float* __restrict__ Out) {
    __shared__ unsigned sA32[64 * 20];
    __shared__ unsigned sB32[64 * 20];
    __bf16* sA = (__bf16*)sA32;
    __bf16* sB = (__bf16*)sB32;

    const int n0 = blockIdx.x * 64;
    const int m0 = blockIdx.y * 64;
    const int t  = threadIdx.x;
    const int w  = t >> 5;
    const int lane = t & 31;
    const int arow = t >> 1;
    const int cseg = (t & 1) << 4;

    f32x8 acc[4] = {};

    for (int k0 = 0; k0 < DIM; k0 += 32) {
        __syncthreads();
        {   // A tile already bf16: two 16B chunks per thread (async when available)
            const __bf16* src = A + (size_t)(m0 + arow) * DIM + k0 + cseg;
            __bf16* dst = &sA[arow * 40 + cseg];
            copy16_g2l(src, dst);
            copy16_g2l(src + 8, dst + 8);
        }
        load_btile(Wb, DIM, k0, n0, sB32, t);
        if (k0 + 32 < DIM)
            __builtin_prefetch(Wb + (size_t)(k0 + 32 + (t >> 3) * 2) * DIM + n0, 0, 1);
        wait_g2l();
        __syncthreads();

        bf16x16 af = load_frag(sA, w * 16, 40, 0);
#pragma unroll
        for (int tt = 0; tt < 4; ++tt) {
            bf16x16 bfrag = load_frag(sB, tt * 16, 40, 0);
            acc[tt] = wmma_bf16(af, bfrag, acc[tt]);
        }
    }

#pragma unroll
    for (int tt = 0; tt < 4; ++tt) {
#pragma unroll
        for (int j = 0; j < 8; ++j) {
            int m   = (w << 4) + j + ((lane >> 4) << 3);
            int col = n0 + (tt << 4) + (lane & 15);
            Out[(size_t)(m0 + m) * DIM + col] = acc[tt][j] + bias[col];
        }
    }
}

// ---------------------------------------------------------------------------
// Windowed cross attention: one WG = (window, head). 128 threads = 4 waves.
// Wave w owns query rows 16w..16w+15. Writes O in place over Q (disjoint slices).
// ---------------------------------------------------------------------------
__global__ __launch_bounds__(128) void attn_win(__bf16* __restrict__ Q,
                                                const __bf16* __restrict__ KV) {
    __shared__ unsigned sQ32[64 * 20];   // 64 tokens x 32 dims, stride 40 halves
    __shared__ unsigned sK32[64 * 20];
    __shared__ unsigned sVt32[32 * 40];  // V transposed [dim][token], stride 80 halves
    __shared__ unsigned sP32[64 * 40];   // probabilities, stride 80 halves
    __bf16* sQ  = (__bf16*)sQ32;
    __bf16* sK  = (__bf16*)sK32;
    __bf16* sVt = (__bf16*)sVt32;
    __bf16* sP  = (__bf16*)sP32;

    const int win  = blockIdx.x;          // 4096 windows
    const int head = blockIdx.y;
    const int b    = win >> 10;
    const int wh   = (win >> 5) & 31;
    const int wwi  = win & 31;
    const int t    = threadIdx.x;
    const int w    = t >> 5;
    const int lane = t & 31;

    auto tok_row = [&](int tok) {         // window_partition/reverse folded in
        int ty = tok >> 3, tx = tok & 7;
        return (((b << 8) + (wh << 3) + ty) << 8) + (wwi << 3) + tx;
    };

    {   // q/k: verbatim 16B chunks (async->LDS when available)
        int row = t >> 2;                 // 0..31
        int c8  = (t & 3) << 3;           // 0,8,16,24
        int g0 = tok_row(row), g1 = tok_row(row + 32);
        copy16_g2l(Q + (size_t)g0 * DIM + head * HD + c8, &sQ[row * 40 + c8]);
        copy16_g2l(Q + (size_t)g1 * DIM + head * HD + c8, &sQ[(row + 32) * 40 + c8]);
        copy16_g2l(KV + (size_t)g0 * (2 * DIM) + head * HD + c8, &sK[row * 40 + c8]);
        copy16_g2l(KV + (size_t)g1 * (2 * DIM) + head * HD + c8, &sK[(row + 32) * 40 + c8]);
    }
    {   // v transposed: pair two adjacent tokens -> packed b32 stores
        int tokp = t & 31;                // token pair (tok = 2*tokp, 2*tokp+1)
        int dseg = (t >> 5) << 3;         // 0,8,16,24
        int g0 = tok_row(2 * tokp);       // tok+1 is the next flat row (tx even)
        const __bf16* v0 = KV + (size_t)g0 * (2 * DIM) + DIM + head * HD + dseg;
        const __bf16* v1 = v0 + 2 * DIM;
        union { uint4 v; unsigned short s[8]; } a, c;
        a.v = *(const uint4*)v0;
        c.v = *(const uint4*)v1;
#pragma unroll
        for (int i = 0; i < 8; ++i)
            sVt32[(dseg + i) * 40 + tokp] =
                (unsigned)a.s[i] | ((unsigned)c.s[i] << 16);
    }
    wait_g2l();
    __syncthreads();

    // S = Q K^T  (16x64 per wave: 4 WMMAs)
    f32x8 sacc[4] = {};
    bf16x16 aq = load_frag(sQ, w * 16, 40, 0);
#pragma unroll
    for (int tt = 0; tt < 4; ++tt) {
        bf16x16 bk = load_frag(sK, tt * 16, 40, 0);
        sacc[tt] = wmma_bf16(aq, bk, sacc[tt]);
    }

    // Softmax over 64 keys. Row m (= j + 8*(lane>=16)) lives in one 16-lane half;
    // xor masks 1,2,4,8 stay inside the half.
    const float scale = 0.17677669529663687f;  // 1/sqrt(32)
#pragma unroll
    for (int j = 0; j < 8; ++j) {
        float x0 = sacc[0][j] * scale, x1 = sacc[1][j] * scale;
        float x2 = sacc[2][j] * scale, x3 = sacc[3][j] * scale;
        float m = fmaxf(fmaxf(x0, x1), fmaxf(x2, x3));
#pragma unroll
        for (int off = 1; off < 16; off <<= 1) m = fmaxf(m, __shfl_xor(m, off, 32));
        float e0 = __expf(x0 - m), e1 = __expf(x1 - m);
        float e2 = __expf(x2 - m), e3 = __expf(x3 - m);
        float s = e0 + e1 + e2 + e3;
#pragma unroll
        for (int off = 1; off < 16; off <<= 1) s += __shfl_xor(s, off, 32);
        float inv = 1.0f / s;
        int row = (w << 4) + j + ((lane >> 4) << 3);
        int c   = lane & 15;
        sP[row * 80 +  0 + c] = (__bf16)(e0 * inv);
        sP[row * 80 + 16 + c] = (__bf16)(e1 * inv);
        sP[row * 80 + 32 + c] = (__bf16)(e2 * inv);
        sP[row * 80 + 48 + c] = (__bf16)(e3 * inv);
    }
    // No WG barrier needed: each wave reads back only its own sP rows.

    // O = P V  (16x32 per wave: 2 N-tiles x 2 K-steps)
    f32x8 oacc[2] = {};
#pragma unroll
    for (int ks = 0; ks < 2; ++ks) {
        bf16x16 ap = load_frag(sP, w * 16, 80, ks * 32);
#pragma unroll
        for (int t2 = 0; t2 < 2; ++t2) {
            bf16x16 bv = load_frag(sVt, t2 * 16, 80, ks * 32);
            oacc[t2] = wmma_bf16(ap, bv, oacc[t2]);
        }
    }

    // Write O in place over Q (disjoint row x head slices per WG)
#pragma unroll
    for (int t2 = 0; t2 < 2; ++t2) {
#pragma unroll
        for (int j = 0; j < 8; ++j) {
            int tok = (w << 4) + j + ((lane >> 4) << 3);
            int d   = (t2 << 4) + (lane & 15);
            int gr  = tok_row(tok);
            Q[(size_t)gr * DIM + head * HD + d] = (__bf16)oacc[t2][j];
        }
    }
}

// ---------------------------------------------------------------------------
// Launch
// ---------------------------------------------------------------------------
extern "C" void kernel_launch(void* const* d_in, const int* in_sizes, int n_in,
                              void* d_out, int out_size, void* d_ws, size_t ws_size,
                              hipStream_t stream) {
    const float* x      = (const float*)d_in[0];
    const float* q_w    = (const float*)d_in[1];
    const float* q_b    = (const float*)d_in[2];
    const float* kv_w   = (const float*)d_in[3];
    const float* kv_b   = (const float*)d_in[4];
    const float* proj_w = (const float*)d_in[5];
    const float* proj_b = (const float*)d_in[6];
    float* out = (float*)d_out;

    // Workspace layout (bf16 elements)
    __bf16* ws    = (__bf16*)d_ws;
    __bf16* wq    = ws;                                   // 384*384
    __bf16* wkv   = wq + 384 * 384;                       // 384*768
    __bf16* wp    = wkv + 384 * 768;                      // 384*384
    __bf16* qbuf  = wp + 384 * 384;                       // M*384 (q, then attn out)
    __bf16* kvbuf = qbuf + (size_t)MROWS * DIM;           // M*768

    const int nq = 384 * 384, nkv = 384 * 768;
    cvt_f32_bf16<<<dim3((nq + 255) / 256), 256, 0, stream>>>(q_w, wq, nq);
    cvt_f32_bf16<<<dim3((nkv + 255) / 256), 256, 0, stream>>>(kv_w, wkv, nkv);
    cvt_f32_bf16<<<dim3((nq + 255) / 256), 256, 0, stream>>>(proj_w, wp, nq);

    // q = x @ q_w + q_b
    gemm_x_bf16<0><<<dim3(DIM / 64, MROWS / 64), 128, 0, stream>>>(x, wq, q_b, qbuf, DIM);
    // kv = roll(x,(-4,-4)) @ kv_w + kv_b
    gemm_x_bf16<1><<<dim3(2 * DIM / 64, MROWS / 64), 128, 0, stream>>>(x, wkv, kv_b, kvbuf, 2 * DIM);
    // windowed attention (output overwrites qbuf)
    attn_win<<<dim3(4096, NHEADS), 128, 0, stream>>>(qbuf, kvbuf);
    // out = attn @ proj_w + proj_b
    gemm_proj<<<dim3(DIM / 64, MROWS / 64), 128, 0, stream>>>(qbuf, wp, proj_b, out);
}
